// DecoderRNNT_70136815944120
// MI455X (gfx1250) — matrix operand order repaired and checked by reading, not compiled, not run on hardware
//
#include <hip/hip_runtime.h>
#include <math.h>

// ---------------------------------------------------------------------------
// RNN-T greedy decoder for MI455X (gfx1250).
//   Kernel 0: prep  -> pack W_dec/W_ih/W_hh/W_out to bf16 pairs in scratch,
//                      fold b_ih+b_hh (halves L2 traffic of the serial chain).
//   Kernel 1: encoder joint projection  hp = h @ W_enc^T + b_enc
//             -> bf16 WMMA (v_wmma_f32_16x16x32_bf16), 2x2 register blocking.
//   Kernel 2: persistent grid-cooperative sequential decode (2048 steps),
//             bf16-packed wave-level matvecs out of L2, LDS-staged operands,
//             online logsumexp/argmax, L2-atomic sense-reversal grid barrier.
// ---------------------------------------------------------------------------

#define TT      2048
#define EPROJS  1024
#define DUNITS  512
#define EMBED   512
#define JOINTD  512
#define ODIM    10000

#define NWG        64
#define NTHREADS   256
#define WAVES_PER_WG (NTHREADS / 32)
#define TOTW       (NWG * WAVES_PER_WG)   // 512 waves

// ---- scratch layout (4-byte words) ----------------------------------------
static constexpr size_t HP_FLOATS = (size_t)TT * JOINTD;        // 1,048,576
static constexpr size_t OFF_Y     = HP_FLOATS;                  // 512
static constexpr size_t OFF_C     = OFF_Y + DUNITS;             // 512
static constexpr size_t OFF_Z     = OFF_C + DUNITS;             // 512
static constexpr size_t OFF_G     = OFF_Z + JOINTD;             // 2048 gates
static constexpr size_t OFF_PMAX  = OFF_G + 4 * DUNITS;         // 64
static constexpr size_t OFF_PSUM  = OFF_PMAX + NWG;             // 64
static constexpr size_t OFF_SCORE = OFF_PSUM + NWG;             // 1
static constexpr size_t OFF_INT   = OFF_SCORE + 4;              // int region
static constexpr size_t INT_WORDS = NWG + 4;  // pidx[NWG], pred, emit, cnt, gen
static constexpr size_t STATE_END = OFF_INT + INT_WORDS;

// bf16-packed weight copies (u32 words, 2 weights per word)
static constexpr size_t OFF_GBIAS = STATE_END;                       // 2048 f32
static constexpr size_t OFF_WDEC  = OFF_GBIAS + 4 * DUNITS;          // 131072
static constexpr size_t OFF_WIH   = OFF_WDEC + (size_t)JOINTD * DUNITS / 2;
static constexpr size_t OFF_WHH   = OFF_WIH + (size_t)4 * DUNITS * EMBED / 2;
static constexpr size_t OFF_WOUT  = OFF_WHH + (size_t)4 * DUNITS * DUNITS / 2;
static constexpr size_t TOTAL_WORDS = OFF_WOUT + (size_t)ODIM * JOINTD / 2;

typedef __attribute__((ext_vector_type(16))) __bf16 v16bf;
typedef __attribute__((ext_vector_type(8)))  float  v8f;

// ===========================================================================
// Kernel 0: pack weights to bf16 pairs, fold gate biases.
// ===========================================================================
__device__ __forceinline__ unsigned pack2_bf16(float a, float b) {
  unsigned ua = __float_as_uint(a), ub = __float_as_uint(b);
  ua = (ua + 0x7fffu + ((ua >> 16) & 1u)) >> 16;          // RNE
  ub = (ub + 0x7fffu + ((ub >> 16) & 1u)) >> 16;
  return ua | (ub << 16);
}

__global__ void __launch_bounds__(256)
prep_weights(const float* __restrict__ W_dec, const float* __restrict__ W_ih,
             const float* __restrict__ W_hh,  const float* __restrict__ W_out,
             const float* __restrict__ b_ih,  const float* __restrict__ b_hh,
             float* __restrict__ ws) {
  unsigned* __restrict__ u = (unsigned*)ws;
  const size_t tid    = (size_t)blockIdx.x * blockDim.x + threadIdx.x;
  const size_t stride = (size_t)gridDim.x * blockDim.x;

  for (size_t i = tid; i < 4 * DUNITS; i += stride)
    ws[OFF_GBIAS + i] = b_ih[i] + b_hh[i];
  for (size_t i = tid; i < (size_t)JOINTD * DUNITS / 2; i += stride)
    u[OFF_WDEC + i] = pack2_bf16(W_dec[2 * i], W_dec[2 * i + 1]);
  for (size_t i = tid; i < (size_t)4 * DUNITS * EMBED / 2; i += stride)
    u[OFF_WIH + i] = pack2_bf16(W_ih[2 * i], W_ih[2 * i + 1]);
  for (size_t i = tid; i < (size_t)4 * DUNITS * DUNITS / 2; i += stride)
    u[OFF_WHH + i] = pack2_bf16(W_hh[2 * i], W_hh[2 * i + 1]);
  for (size_t i = tid; i < (size_t)ODIM * JOINTD / 2; i += stride)
    u[OFF_WOUT + i] = pack2_bf16(W_out[2 * i], W_out[2 * i + 1]);
}

// ===========================================================================
// Kernel 1: hp = h @ W_enc^T + b_enc via bf16 WMMA, 2x2 tile blocking.
// Each wave computes a 32x32 output block (4 accumulators, 4 WMMAs/K-step,
// each A/B fragment reused twice).
// grid = (JOINTD/32, TT/32/4), block = 128 (4 waves).
// ===========================================================================
__device__ __forceinline__ v16bf load_a_frag(const float* __restrict__ Arow,
                                             int k0, int kHiA) {
  v16bf a;
#pragma unroll
  for (int e = 0; e < 16; ++e) a[e] = (__bf16)Arow[k0 + e + kHiA];
  return a;
}
__device__ __forceinline__ v16bf load_b_frag(const float* __restrict__ Brow,
                                             int k0, int kHiB) {
  v16bf b;
#pragma unroll
  for (int e = 0; e < 16; ++e) b[e] = (__bf16)Brow[k0 + e + kHiB];
  return b;
}

__global__ void __launch_bounds__(128)
henc_proj_wmma(const float* __restrict__ h,
               const float* __restrict__ Wenc,
               const float* __restrict__ benc,
               float* __restrict__ hp) {
  const int lane   = threadIdx.x & 31;
  const int wave   = threadIdx.x >> 5;
  const int tileM0 = (blockIdx.y * 4 + wave) * 2;   // 0..126, step 2
  const int tileN0 = blockIdx.x * 2;                // 0..30,  step 2

  // A fragment (16-bit A 16x32): row m = lane&15, elem e -> k = e+16*(lane>=16)
  const int mA   = lane & 15;
  const int kHiA = (lane & 16);
  // B fragment (16-bit B 32x16): col n = lane&15, elem e -> k = e+16*(lane>=16)
  const int nB   = lane & 15;
  const int kHiB = (lane & 16);

  const float* __restrict__ A0 = h + (size_t)(tileM0 * 16 + mA) * EPROJS;
  const float* __restrict__ A1 = A0 + (size_t)16 * EPROJS;
  const float* __restrict__ B0 = Wenc + (size_t)(tileN0 * 16 + nB) * EPROJS;
  const float* __restrict__ B1 = B0 + (size_t)16 * EPROJS;

  v8f acc00 = {}, acc01 = {}, acc10 = {}, acc11 = {};
  for (int k0 = 0; k0 < EPROJS; k0 += 32) {
    const v16bf a0 = load_a_frag(A0, k0, kHiA);
    const v16bf a1 = load_a_frag(A1, k0, kHiA);
    const v16bf b0 = load_b_frag(B0, k0, kHiB);
    const v16bf b1 = load_b_frag(B1, k0, kHiB);
    acc00 = __builtin_amdgcn_wmma_f32_16x16x32_bf16(false, a0, false, b0,
                                                    (short)0, acc00, false, false);
    acc01 = __builtin_amdgcn_wmma_f32_16x16x32_bf16(false, a0, false, b1,
                                                    (short)0, acc01, false, false);
    acc10 = __builtin_amdgcn_wmma_f32_16x16x32_bf16(false, a1, false, b0,
                                                    (short)0, acc10, false, false);
    acc11 = __builtin_amdgcn_wmma_f32_16x16x32_bf16(false, a1, false, b1,
                                                    (short)0, acc11, false, false);
  }

  // C/D layout: VGPR v -> m = v + 8*(lane>=16); n = lane&15
  const int   mBase = (lane >> 4) * 8;
  const float bias0 = benc[tileN0 * 16 + nB];
  const float bias1 = benc[(tileN0 + 1) * 16 + nB];
#pragma unroll
  for (int v = 0; v < 8; ++v) {
    const int m0 = tileM0 * 16 + mBase + v;
    const int m1 = m0 + 16;
    float* r0 = hp + (size_t)m0 * JOINTD + tileN0 * 16 + nB;
    float* r1 = hp + (size_t)m1 * JOINTD + tileN0 * 16 + nB;
    r0[0]  = acc00[v] + bias0;
    r0[16] = acc01[v] + bias1;
    r1[0]  = acc10[v] + bias0;
    r1[16] = acc11[v] + bias1;
  }
}

// ===========================================================================
// Decode helpers
// ===========================================================================
__device__ __forceinline__ float wave_reduce_sum(float s) {
#pragma unroll
  for (int off = 16; off >= 1; off >>= 1) s += __shfl_xor(s, off, 32);
  return s;
}

__device__ __forceinline__ float sigmoidf(float x) {
  return 1.0f / (1.0f + __expf(-x));
}

// 512-dot: bf16-packed weight row (256 u32) x fp32 vector (LDS-staged)
__device__ __forceinline__ float dot_bf16(const unsigned* __restrict__ w,
                                          const float* __restrict__ v, int lane) {
  float s = 0.f;
#pragma unroll
  for (int k = lane; k < 256; k += 32) {
    const unsigned u = w[k];
    s = fmaf(__uint_as_float(u << 16), v[2 * k], s);
    s = fmaf(__uint_as_float(u & 0xffff0000u), v[2 * k + 1], s);
  }
  return wave_reduce_sum(s);
}

__device__ __forceinline__ void grid_barrier(unsigned* cnt, unsigned* gen) {
  __threadfence();
  __syncthreads();
  if (threadIdx.x == 0) {
    unsigned g = __hip_atomic_load(gen, __ATOMIC_RELAXED, __HIP_MEMORY_SCOPE_AGENT);
    unsigned arrived =
        __hip_atomic_fetch_add(cnt, 1u, __ATOMIC_ACQ_REL, __HIP_MEMORY_SCOPE_AGENT);
    if (arrived == NWG - 1) {
      __hip_atomic_store(cnt, 0u, __ATOMIC_RELAXED, __HIP_MEMORY_SCOPE_AGENT);
      __hip_atomic_fetch_add(gen, 1u, __ATOMIC_RELEASE, __HIP_MEMORY_SCOPE_AGENT);
    } else {
      while (__hip_atomic_load(gen, __ATOMIC_ACQUIRE, __HIP_MEMORY_SCOPE_AGENT) == g) {
        __builtin_amdgcn_s_sleep(2);
      }
    }
  }
  __syncthreads();
  __threadfence();
}

// ===========================================================================
// Kernel 2: persistent grid-cooperative greedy decode.
// ===========================================================================
__global__ void __launch_bounds__(NTHREADS, 1)
rnnt_decode(const float* __restrict__ embed,
            const float* __restrict__ b_out,
            float* __restrict__ ws, float* __restrict__ dout) {
  float* __restrict__ hp    = ws;
  float* __restrict__ y     = ws + OFF_Y;
  float* __restrict__ c     = ws + OFF_C;
  float* __restrict__ z     = ws + OFF_Z;
  float* __restrict__ gates = ws + OFF_G;
  float* __restrict__ pmax  = ws + OFF_PMAX;
  float* __restrict__ psum  = ws + OFF_PSUM;
  float* __restrict__ score = ws + OFF_SCORE;
  const float* __restrict__ gbias = ws + OFF_GBIAS;
  const unsigned* __restrict__ wdec = (const unsigned*)ws + OFF_WDEC;
  const unsigned* __restrict__ wih  = (const unsigned*)ws + OFF_WIH;
  const unsigned* __restrict__ whh  = (const unsigned*)ws + OFF_WHH;
  const unsigned* __restrict__ wout = (const unsigned*)ws + OFF_WOUT;
  int*      ip        = (int*)(ws + OFF_INT);
  int*      pidx      = ip;
  int*      ctrl_pred = ip + NWG;
  int*      ctrl_emit = ip + NWG + 1;
  unsigned* bar_cnt   = (unsigned*)(ip + NWG + 2);
  unsigned* bar_gen   = (unsigned*)(ip + NWG + 3);

  const int wgid  = blockIdx.x;
  const int tid   = threadIdx.x;
  const int lane  = tid & 31;
  const int wv    = tid >> 5;
  const int gwave = wgid * WAVES_PER_WG + wv;
  const int gtid  = wgid * NTHREADS + tid;

  __shared__ float smax[WAVES_PER_WG];
  __shared__ float ssum[WAVES_PER_WG];
  __shared__ int   sidx[WAVES_PER_WG];
  __shared__ float sh_y[DUNITS];    // staged y (also used in Phase D)
  __shared__ float sh_v[JOINTD];    // staged z (Phase B) / ey (Phase D)

  // ---- initial prediction-net step: zero input, zero state ---------------
  for (int d = gtid; d < DUNITS; d += NWG * NTHREADS) {
    float ig = sigmoidf(gbias[d]);
    float gg = tanhf(gbias[2 * DUNITS + d]);
    float og = sigmoidf(gbias[3 * DUNITS + d]);
    float cc = ig * gg;                 // f*0 + i*tanh(g)
    c[d] = cc;
    y[d] = og * tanhf(cc);
  }
  grid_barrier(bar_cnt, bar_gen);

  // ---- time loop ---------------------------------------------------------
  for (int t = 0; t < TT; ++t) {
    // Phase A: z = tanh(hp[t] + W_dec @ y)
    for (int d = tid; d < DUNITS; d += NTHREADS) sh_y[d] = y[d];
    __syncthreads();
    for (int r = gwave; r < JOINTD; r += TOTW) {
      float s = dot_bf16(wdec + (size_t)r * 256, sh_y, lane);
      if (lane == 0) z[r] = tanhf(hp[(size_t)t * JOINTD + r] + s);
    }
    grid_barrier(bar_cnt, bar_gen);

    // Phase B: logits = W_out @ z + b_out, fused online max/argmax/sumexp.
    for (int d = tid; d < JOINTD; d += NTHREADS) sh_v[d] = z[d];
    __syncthreads();
    float m = -INFINITY, ssacc = 0.f;
    int   bi = 0;
    for (int r = gwave; r < ODIM; r += TOTW) {
      float s = dot_bf16(wout + (size_t)r * 256, sh_v, lane) + b_out[r];
      if (s > m) { ssacc = ssacc * __expf(m - s) + 1.f; m = s; bi = r; }
      else       { ssacc += __expf(s - m); }
    }
    if (lane == 0) { smax[wv] = m; ssum[wv] = ssacc; sidx[wv] = bi; }
    __syncthreads();
    if (tid == 0) {
      float M = -INFINITY, S = 0.f; int I = 0;
      for (int w = 0; w < WAVES_PER_WG; ++w) {
        float m2 = smax[w], s2 = ssum[w];
        if (m2 > M) { S = S * __expf(M - m2) + s2; M = m2; I = sidx[w]; }
        else        { S += s2 * __expf(m2 - M); }
      }
      pmax[wgid] = M; psum[wgid] = S; pidx[wgid] = I;
    }
    grid_barrier(bar_cnt, bar_gen);

    // Phase C: final merge, token emit, score update (single thread).
    if (wgid == 0 && tid == 0) {
      float M = -INFINITY, S = 0.f; int I = 0;
      for (int w = 0; w < NWG; ++w) {
        float m2 = pmax[w], s2 = psum[w];
        if (m2 > M) { S = S * __expf(M - m2) + s2; M = m2; I = pidx[w]; }
        else        { S += s2 * __expf(m2 - M); }
      }
      const int emitted = (I != 0);
      const float logp = -__logf(S);     // max(log_softmax) = -log(sumexp)
      if (emitted) score[0] += logp;
      dout[t] = emitted ? (float)I : 0.0f;
      *ctrl_pred = I;
      *ctrl_emit = emitted;
    }
    grid_barrier(bar_cnt, bar_gen);

    const int pred    = *ctrl_pred;
    const int emitted = *ctrl_emit;      // uniform across the grid

    if (emitted) {
      // Phase D: gates = W_ih@ey + W_hh@y + (b_ih+b_hh); sh_y still holds y.
      const float* __restrict__ ey = embed + (size_t)pred * EMBED;
      for (int d = tid; d < EMBED; d += NTHREADS) sh_v[d] = ey[d];
      __syncthreads();
      for (int g = gwave; g < 4 * DUNITS; g += TOTW) {
        const unsigned* __restrict__ wi = wih + (size_t)g * 256;
        const unsigned* __restrict__ wh = whh + (size_t)g * 256;
        float s = 0.f;
#pragma unroll
        for (int k = lane; k < 256; k += 32) {
          const unsigned ui = wi[k], uh = wh[k];
          s = fmaf(__uint_as_float(ui << 16),          sh_v[2 * k],     s);
          s = fmaf(__uint_as_float(ui & 0xffff0000u),  sh_v[2 * k + 1], s);
          s = fmaf(__uint_as_float(uh << 16),          sh_y[2 * k],     s);
          s = fmaf(__uint_as_float(uh & 0xffff0000u),  sh_y[2 * k + 1], s);
        }
        s = wave_reduce_sum(s);
        if (lane == 0) gates[g] = s + gbias[g];
      }
      grid_barrier(bar_cnt, bar_gen);

      // Phase E: elementwise cell update.
      for (int d = gtid; d < DUNITS; d += NWG * NTHREADS) {
        float ig = sigmoidf(gates[d]);
        float fg = sigmoidf(gates[DUNITS + d]);
        float gg = tanhf(gates[2 * DUNITS + d]);
        float og = sigmoidf(gates[3 * DUNITS + d]);
        float cc = fg * c[d] + ig * gg;
        c[d] = cc;
        y[d] = og * tanhf(cc);
      }
      grid_barrier(bar_cnt, bar_gen);
    }
  }

  if (wgid == 0 && tid == 0) dout[TT] = score[0];
}

// ===========================================================================
extern "C" void kernel_launch(void* const* d_in, const int* in_sizes, int n_in,
                              void* d_out, int out_size, void* d_ws, size_t ws_size,
                              hipStream_t stream) {
  const float* h     = (const float*)d_in[0];
  const float* embed = (const float*)d_in[1];
  const float* W_ih  = (const float*)d_in[2];
  const float* W_hh  = (const float*)d_in[3];
  const float* b_ih  = (const float*)d_in[4];
  const float* b_hh  = (const float*)d_in[5];
  const float* W_enc = (const float*)d_in[6];
  const float* b_enc = (const float*)d_in[7];
  const float* W_dec = (const float*)d_in[8];
  const float* W_out = (const float*)d_in[9];
  const float* b_out = (const float*)d_in[10];
  float* out = (float*)d_out;
  float* ws  = (float*)d_ws;

  // Zero the state/control region. hp and bf16 copies are fully rewritten.
  hipMemsetAsync(ws + OFF_Y, 0, (STATE_END - OFF_Y) * sizeof(float), stream);

  // Kernel 0: bf16-pack the serial-chain weights (one-time per launch).
  prep_weights<<<2048, 256, 0, stream>>>(W_dec, W_ih, W_hh, W_out,
                                         b_ih, b_hh, ws);

  // Kernel 1: encoder projection via bf16 WMMA (2x2 blocked).
  dim3 gridP(JOINTD / 32, TT / 32 / 4);   // (16, 16)
  henc_proj_wmma<<<gridP, 128, 0, stream>>>(h, W_enc, b_enc, ws);

  // Kernel 2: persistent grid-cooperative greedy decode.
  rnnt_decode<<<NWG, NTHREADS, 0, stream>>>(embed, b_out, ws, out);
}
